// RelativeMultiheadAttention_46420006535655
// MI455X (gfx1250) — matrix operand order repaired and checked by reading, not compiled, not run on hardware
//
#include <hip/hip_runtime.h>
#include <hip/hip_bf16.h>

// Problem dims (fixed by reference)
#define Bq 4
#define Tq 1024
#define Dq 1024
#define Hq 16
#define DHq 64

typedef __attribute__((ext_vector_type(16))) _Float16 v16h;
typedef __attribute__((ext_vector_type(8)))  float    v8f;
typedef __attribute__((ext_vector_type(4)))  unsigned int v4u;
typedef __attribute__((ext_vector_type(8)))  int      v8i;
typedef __attribute__((ext_vector_type(4)))  int      v4i;

// Attention-kernel LDS layout (dynamic shared, offset 0 = wave LDS base)
#define SC_OFF   0                      // 16*1024 f32 scores      (64 KB)
#define PR_OFF   (16 * Tq * 4)          // 16*1024 f16 probs       (32 KB)
#define RED_OFF  (PR_OFF + 16 * Tq * 2) // 256 f32 reduction       ( 1 KB)
#define V_OFF    (RED_OFF + 256 * 4)    // 1024*64 f16 V head tile (128 KB)
#define SMEM_BYTES (V_OFF + Tq * DHq * 2)

// ---------------------------------------------------------------------------
// WMMA wrapper: D(f32 16x16) = A(f16 16x32) * B(f16 32x16) + C
// ---------------------------------------------------------------------------
__device__ __forceinline__ v8f wmma_f16f32(v16h a, v16h b, v8f c) {
  return __builtin_amdgcn_wmma_f32_16x16x32_f16(
      /*neg_a=*/false, a, /*neg_b=*/false, b,
      /*c_mod=*/(short)0, c, /*reuse_a=*/false, /*reuse_b=*/false);
}

// ---------------------------------------------------------------------------
// A fragment (16x32 f16, row-major source, leading dim ld).
// ISA 7.12.2: lane m=lane&15, half=lane>>4; VGPR0..3 hold K = 8*half + 0..7,
// VGPR4..7 hold K = 16 + 8*half + 0..7. Two contiguous 16B chunks per lane.
// ---------------------------------------------------------------------------
__device__ __forceinline__ v16h load_A_frag(const _Float16* p, int ld,
                                            int row0, int col0) {
  int lane = threadIdx.x & 31;
  int m = lane & 15, hh = lane >> 4;
  const _Float16* base = p + (size_t)(row0 + m) * ld + col0 + 8 * hh;
  v16h a;
#pragma unroll
  for (int j = 0; j < 4; ++j) { a[2*j]   = base[2*j];      a[2*j+1] = base[2*j+1]; }
#pragma unroll
  for (int j = 0; j < 4; ++j) { a[8+2*j] = base[16 + 2*j]; a[9+2*j] = base[17 + 2*j]; }
  return a;
}

// ---------------------------------------------------------------------------
// B fragment (32x16 f16): lane holds n=lane&15, half=lane>>4; elements 0..7 hold
// K = 8*half + 0..7, elements 8..15 hold K = 16 + 8*half + 0..7 (row-major source).
// ---------------------------------------------------------------------------
__device__ __forceinline__ v16h load_B_rowmajor(const _Float16* p, int ld,
                                                int k0, int n0) {
  int lane = threadIdx.x & 31;
  int n = lane & 15, hh = lane >> 4;
  const _Float16* base = p + (size_t)(k0 + 8 * hh) * ld + n0 + n;
  v16h b;
#pragma unroll
  for (int j = 0; j < 8; ++j) { b[j] = base[(size_t)j * ld]; b[8+j] = base[(size_t)(16 + j) * ld]; }
  return b;
}

// ---------------------------------------------------------------------------
// fp32 -> fp16 elementwise convert
// ---------------------------------------------------------------------------
__global__ void f32_to_f16_kernel(const float* __restrict__ x,
                                  _Float16* __restrict__ y, int n) {
  int i = blockIdx.x * blockDim.x + threadIdx.x;
  if (i < n) y[i] = (_Float16)x[i];
}

// ---------------------------------------------------------------------------
// Per-head transpose: src (B*T, D) row-major -> dst (B*H*DH, T) row-major,
// dst[((bb*H+h)*DH + d)*T + s] = src[(bb*T+s)*D + h*DH + d].
// 16x16 LDS tiles, conflict-padded.
// ---------------------------------------------------------------------------
__global__ void transpose_head_kernel(const _Float16* __restrict__ src,
                                      _Float16* __restrict__ dst) {
  __shared__ _Float16 tile[16][17];
  int blk = blockIdx.x;
  int s0 = (blk % (Tq / 16)) * 16;  blk /= (Tq / 16);
  int d0 = (blk % (DHq / 16)) * 16; blk /= (DHq / 16);
  int h  = blk % Hq;
  int bb = blk / Hq;
  int tx = threadIdx.x & 15, ty = threadIdx.x >> 4;
  tile[ty][tx] = src[(size_t)(bb * Tq + s0 + ty) * Dq + h * DHq + d0 + tx];
  __syncthreads();
  dst[(size_t)((bb * Hq + h) * DHq + d0 + ty) * Tq + s0 + tx] = tile[tx][ty];
}

// ---------------------------------------------------------------------------
// Row LayerNorm (row length 1024) + fp16 convert. One block (256t) per row.
// ---------------------------------------------------------------------------
__global__ void ln_f16_kernel(const float* __restrict__ x,
                              const float* __restrict__ g,
                              const float* __restrict__ beta,
                              _Float16* __restrict__ y) {
  __shared__ float red[256];
  int row = blockIdx.x;
  int tid = threadIdx.x;
  const float* xr = x + (size_t)row * Dq;
  float v[4];
  float s = 0.f;
#pragma unroll
  for (int i = 0; i < 4; ++i) { v[i] = xr[tid + i * 256]; s += v[i]; }
  red[tid] = s; __syncthreads();
  for (int off = 128; off > 0; off >>= 1) {
    if (tid < off) red[tid] += red[tid + off];
    __syncthreads();
  }
  float mu = red[0] * (1.0f / Dq);
  __syncthreads();
  float s2 = 0.f;
#pragma unroll
  for (int i = 0; i < 4; ++i) { float d = v[i] - mu; s2 += d * d; }
  red[tid] = s2; __syncthreads();
  for (int off = 128; off > 0; off >>= 1) {
    if (tid < off) red[tid] += red[tid + off];
    __syncthreads();
  }
  float rstd = rsqrtf(red[0] * (1.0f / Dq) + 1e-5f);
  _Float16* yr = y + (size_t)row * Dq;
#pragma unroll
  for (int i = 0; i < 4; ++i) {
    int c = tid + i * 256;
    yr[c] = (_Float16)((v[i] - mu) * rstd * g[c] + beta[c]);
  }
}

// ---------------------------------------------------------------------------
// WMMA GEMM: C = alpha * (A(f16 MxK) @ B(f16 KxN) + bias[n])
// Each wave computes a 16x64 output strip (A fragment reused across 4 B tiles).
// ---------------------------------------------------------------------------
__global__ void gemm_wmma_f16(const _Float16* __restrict__ A,
                              const _Float16* __restrict__ Bw,
                              const float* __restrict__ bias,
                              _Float16* __restrict__ Cf16,
                              float* __restrict__ Cf32,
                              int M, int N, int K, float alpha) {
  int wid = blockIdx.x * (blockDim.x >> 5) + (threadIdx.x >> 5);
  int ntile = N >> 6;
  int total = (M >> 4) * ntile;
  if (wid >= total) return;
  int tm = wid / ntile;
  int tn = wid % ntile;
  int lane = threadIdx.x & 31, hh = lane >> 4, nn = lane & 15;

  v8f zf = {};
  v8f acc[4] = { zf, zf, zf, zf };
  for (int k0 = 0; k0 < K; k0 += 32) {
    v16h a = load_A_frag(A, K, tm * 16, k0);
#pragma unroll
    for (int t = 0; t < 4; ++t) {
      v16h b = load_B_rowmajor(Bw, N, k0, tn * 64 + t * 16);
      acc[t] = wmma_f16f32(a, b, acc[t]);
    }
  }
#pragma unroll
  for (int t = 0; t < 4; ++t) {
#pragma unroll
    for (int i = 0; i < 8; ++i) {
      int gm = tm * 16 + i + 8 * hh;
      int gn = tn * 64 + t * 16 + nn;
      float v = acc[t][i] + (bias ? bias[gn] : 0.0f);
      v *= alpha;
      if (Cf32) Cf32[(size_t)gm * N + gn] = v;
      else      Cf16[(size_t)gm * N + gn] = (_Float16)v;
    }
  }
}

// ---------------------------------------------------------------------------
// Issue a TDM tensor_load_to_lds for one head of V:
// 2D tile 64 (d) x 1024 (s) of f16, global row stride D elems, dense in LDS.
// Descriptor fields per ISA 8.3 / 8.4. All operands are wave-uniform SGPRs.
// ---------------------------------------------------------------------------
__device__ __forceinline__ void tdm_load_v_head(const _Float16* vhead_global,
                                                unsigned lds_byte_off) {
  unsigned long long ga = (unsigned long long)(uintptr_t)vhead_global;
  v4u g0;
  g0[0] = 1u;                                  // count=1, user descriptor
  g0[1] = lds_byte_off;                        // lds_addr
  g0[2] = (unsigned)(ga & 0xFFFFFFFFu);        // global_addr[31:0]
  g0[3] = (unsigned)((ga >> 32) & 0x1FFFFFFu)  // global_addr[56:32]
          | (2u << 30);                        // type = 2 ("image")
  v8i g1;
  g1[0] = (1 << 16);            // workgroup_mask=0, data_size=1 (2 bytes)
  g1[1] = (int)(64u << 16);     // tensor_dim0[15:0] = 64   (bits 63:48)
  g1[2] = (int)(1024u << 16);   // tensor_dim1[15:0] = 1024 (bits 95:80)
  g1[3] = (int)(64u << 16);     // tile_dim0 = 64           (bits 127:112)
  g1[4] = 1024;                 // tile_dim1 = 1024         (bits 143:128)
  g1[5] = 1024;                 // tensor_dim0_stride[31:0] = 1024 elems
  g1[6] = 0;
  g1[7] = 0;
  v4i g2 = { 0, 0, 0, 0 };      // 2D tensor: dims 2/3 unused
  v4i g3 = { 0, 0, 0, 0 };
#if defined(__clang_major__) && (__clang_major__ >= 23)
  v8i gx = { 0, 0, 0, 0, 0, 0, 0, 0 };
  __builtin_amdgcn_tensor_load_to_lds(g0, g1, g2, g3, gx, 0);
#else
  __builtin_amdgcn_tensor_load_to_lds(g0, g1, g2, g3, 0);
#endif
}

// ---------------------------------------------------------------------------
// Fused attention. One block = 16 query rows of one (b,h); 8 waves.
//  - wave 0 kicks a TDM DMA of the whole V head into LDS (overlaps scores)
//  - scores = Q*K^T + Q*(0.1R)^T via WMMA (K/R pre-transposed)
//  - mask -> softmax -> probs fp32 to d_out + fp16 to LDS
//  - PV: 8 waves (4 d-tiles x 2 s-halves) on LDS probs x LDS V, reduce in LDS
// ---------------------------------------------------------------------------
__global__ void attn_softmax_pv(const _Float16* __restrict__ Q16,
                                const _Float16* __restrict__ Kt16,
                                const _Float16* __restrict__ Vp16,
                                const _Float16* __restrict__ Rt16,
                                const int* __restrict__ mask,
                                float* __restrict__ probs_out,
                                _Float16* __restrict__ AO16) {
  extern __shared__ char smem[];
  float*    s_scores = (float*)(smem + SC_OFF);    // 16 x 1024 f32
  _Float16* s_probs  = (_Float16*)(smem + PR_OFF); // 16 x 1024 f16
  float*    s_red    = (float*)(smem + RED_OFF);   // 256 f32
  _Float16* s_v      = (_Float16*)(smem + V_OFF);  // 1024 x 64 f16 (V head)
  float*    s_part   = s_scores;                   // PV partials alias scores

  int blk = blockIdx.x;
  int t0  = (blk % (Tq / 16)) * 16;
  int h   = (blk / (Tq / 16)) % Hq;
  int bb  = blk / ((Tq / 16) * Hq);

  int tid  = threadIdx.x;
  int wid  = tid >> 5;
  int lane = tid & 31;
  int hh   = lane >> 4;
  int nn   = lane & 15;

  const _Float16* Qbase  = Q16 + (size_t)(bb * Tq + t0) * Dq + h * DHq;
  const _Float16* KtHead = Kt16 + (size_t)((bb * Hq + h) * DHq) * Tq; // DH x T
  const _Float16* RtHead = Rt16 + (size_t)((bb * Hq + h) * DHq) * Tq; // DH x T
  const _Float16* Vhead  = Vp16 + (size_t)(bb * Tq) * Dq + h * DHq;   // T x DH (ld=D)

  // Kick the async V-head DMA (tracked by TENSORcnt on wave 0 only).
  if (wid == 0) tdm_load_v_head(Vhead, (unsigned)V_OFF);

  // Q A-fragments for d = 0..31 and 32..63 (reused for all 64 s-tiles)
  v16h a0 = load_A_frag(Qbase, Dq, 0, 0);
  v16h a1 = load_A_frag(Qbase, Dq, 0, 32);

  // ---- score phase: 8 s-tiles per wave, 4 chained WMMAs per tile ----
  for (int st = wid; st < Tq / 16; st += 8) {
    int s0 = st * 16;
    v8f acc = {};
    v16h bk0 = load_B_rowmajor(KtHead, Tq, 0,  s0);   // rows d=0..31
    acc = wmma_f16f32(a0, bk0, acc);
    v16h bk1 = load_B_rowmajor(KtHead, Tq, 32, s0);   // rows d=32..63
    acc = wmma_f16f32(a1, bk1, acc);
    v16h br0 = load_B_rowmajor(RtHead, Tq, 0,  s0);
    acc = wmma_f16f32(a0, br0, acc);
    v16h br1 = load_B_rowmajor(RtHead, Tq, 32, s0);
    acc = wmma_f16f32(a1, br1, acc);

    int s  = s0 + nn;
    int mk = mask[bb * Tq + s];
#pragma unroll
    for (int i = 0; i < 8; ++i) {
      int m = i + 8 * hh;
      float v = acc[i];
      if (mk == 0) v = -1.0e9f;
      s_scores[m * Tq + s] = v;
    }
  }
  __syncthreads();

  // ---- softmax: 16 threads per row ----
  int row = tid >> 4;
  int c   = tid & 15;
  float lmax = -3.4e38f;
  for (int s = c; s < Tq; s += 16) lmax = fmaxf(lmax, s_scores[row * Tq + s]);
  s_red[row * 16 + c] = lmax;
  __syncthreads();
  float rmax = -3.4e38f;
#pragma unroll
  for (int j = 0; j < 16; ++j) rmax = fmaxf(rmax, s_red[row * 16 + j]);
  __syncthreads();

  float lsum = 0.f;
  for (int s = c; s < Tq; s += 16) {
    float p = __expf(s_scores[row * Tq + s] - rmax);
    s_scores[row * Tq + s] = p;
    lsum += p;
  }
  s_red[row * 16 + c] = lsum;
  __syncthreads();
  float rsum = 0.f;
#pragma unroll
  for (int j = 0; j < 16; ++j) rsum += s_red[row * 16 + j];
  float inv = 1.0f / rsum;

  float* prow = probs_out + ((size_t)((bb * Hq + h) * Tq) + t0 + row) * Tq;
  for (int s = c; s < Tq; s += 16) {
    float p = s_scores[row * Tq + s] * inv;
    prow[s] = p;                        // fp32 probs -> d_out (the 256MB stream)
    s_probs[row * Tq + s] = (_Float16)p;
  }

  // Drain the V DMA on the issuing wave, then make it visible to all waves.
  if (wid == 0) __builtin_amdgcn_s_wait_tensorcnt(0);
  __syncthreads();

  // ---- PV: wave w -> d-tile (w&3), s-half (w>>2); K=512 per wave ----
  {
    int dt = wid & 3;
    int sh = wid >> 2;
    v8f acc = {};
    for (int s0 = sh * (Tq / 2); s0 < (sh + 1) * (Tq / 2); s0 += 32) {
      v16h a  = load_A_frag(s_probs, Tq, 0, s0);           // LDS
      v16h bv = load_B_rowmajor(s_v, DHq, s0, dt * 16);    // LDS (TDM-staged)
      acc = wmma_f16f32(a, bv, acc);
    }
#pragma unroll
    for (int i = 0; i < 8; ++i) {
      int m = i + 8 * hh;
      s_part[((wid * 16) + m) * 16 + nn] = acc[i];   // [sh*4+dt][m][n]
    }
  }
  __syncthreads();

  // combine the two s-half partials, write attn_out f16
  {
    // 4 dt * 16 m * 16 n = 1024 elements; 4 per thread
    for (int e = tid; e < 1024; e += 256) {
      int n  = e & 15;
      int m  = (e >> 4) & 15;
      int dt = e >> 8;
      float v = s_part[((dt * 16) + m) * 16 + n] +
                s_part[(((4 + dt) * 16) + m) * 16 + n];
      AO16[(size_t)(bb * Tq + t0 + m) * Dq + h * DHq + dt * 16 + n] = (_Float16)v;
    }
  }
}

// ---------------------------------------------------------------------------
// Host-side orchestration
// ---------------------------------------------------------------------------
extern "C" void kernel_launch(void* const* d_in, const int* in_sizes, int n_in,
                              void* d_out, int out_size, void* d_ws, size_t ws_size,
                              hipStream_t stream) {
  (void)in_sizes; (void)n_in; (void)out_size; (void)ws_size;
  const float* query = (const float*)d_in[0];
  const float* key   = (const float*)d_in[1];
  const float* value = (const float*)d_in[2];
  const float* relp  = (const float*)d_in[3];
  const int*   mask  = (const int*)d_in[4];
  const float* qn_g  = (const float*)d_in[5];
  const float* qn_b  = (const float*)d_in[6];
  const float* kn_g  = (const float*)d_in[7];
  const float* kn_b  = (const float*)d_in[8];
  const float* vn_g  = (const float*)d_in[9];
  const float* vn_b  = (const float*)d_in[10];
  const float* Wq    = (const float*)d_in[11];
  const float* bq    = (const float*)d_in[12];
  const float* Wk    = (const float*)d_in[13];
  const float* bk    = (const float*)d_in[14];
  const float* Wv    = (const float*)d_in[15];
  const float* bv    = (const float*)d_in[16];
  const float* Wr    = (const float*)d_in[17];
  const float* Wo    = (const float*)d_in[18];
  const float* bo    = (const float*)d_in[19];

  const int MT = Bq * Tq;                 // 4096 rows
  const size_t WW = (size_t)Dq * Dq;      // 1M weight elems
  const size_t XX = (size_t)MT * Dq;      // 4M activation elems

  char* ws = (char*)d_ws;
  size_t off = 0;
  auto alloc = [&](size_t bytes) -> void* {
    void* p = ws + off;
    off += (bytes + 255) & ~(size_t)255;
    return p;
  };
  _Float16* Wq16 = (_Float16*)alloc(WW * 2);
  _Float16* Wk16 = (_Float16*)alloc(WW * 2);
  _Float16* Wv16 = (_Float16*)alloc(WW * 2);
  _Float16* Wr16 = (_Float16*)alloc(WW * 2);
  _Float16* Wo16 = (_Float16*)alloc(WW * 2);
  _Float16* Xq16 = (_Float16*)alloc(XX * 2);
  _Float16* Xk16 = (_Float16*)alloc(XX * 2);
  _Float16* Xv16 = (_Float16*)alloc(XX * 2);
  _Float16* Xr16 = (_Float16*)alloc(XX * 2);
  _Float16* Q16  = (_Float16*)alloc(XX * 2);
  _Float16* Kp16 = (_Float16*)alloc(XX * 2);
  _Float16* Vp16 = (_Float16*)alloc(XX * 2);
  _Float16* Rp16 = (_Float16*)alloc(XX * 2);
  _Float16* Kt16 = (_Float16*)alloc(XX * 2);  // per-head transposed K
  _Float16* Rt16 = (_Float16*)alloc(XX * 2);  // per-head transposed 0.1*R
  _Float16* AO16 = (_Float16*)alloc(XX * 2);

  // 1) weights + rel_pos -> fp16
  f32_to_f16_kernel<<<(int)(WW / 256), 256, 0, stream>>>(Wq, Wq16, (int)WW);
  f32_to_f16_kernel<<<(int)(WW / 256), 256, 0, stream>>>(Wk, Wk16, (int)WW);
  f32_to_f16_kernel<<<(int)(WW / 256), 256, 0, stream>>>(Wv, Wv16, (int)WW);
  f32_to_f16_kernel<<<(int)(WW / 256), 256, 0, stream>>>(Wr, Wr16, (int)WW);
  f32_to_f16_kernel<<<(int)(WW / 256), 256, 0, stream>>>(Wo, Wo16, (int)WW);
  f32_to_f16_kernel<<<(int)(XX / 256), 256, 0, stream>>>(relp, Xr16, (int)XX);

  // 2) LayerNorm -> fp16
  ln_f16_kernel<<<MT, 256, 0, stream>>>(query, qn_g, qn_b, Xq16);
  ln_f16_kernel<<<MT, 256, 0, stream>>>(key,   kn_g, kn_b, Xk16);
  ln_f16_kernel<<<MT, 256, 0, stream>>>(value, vn_g, vn_b, Xv16);

  // 3) projection GEMMs (4096x1024 @ 1024x1024), WMMA f16 -> f16
  const int gemm_blocks = ((MT / 16) * (Dq / 64)) / 8;   // 512 blocks of 8 waves
  gemm_wmma_f16<<<gemm_blocks, 256, 0, stream>>>(Xq16, Wq16, bq, Q16, nullptr,
                                                 MT, Dq, Dq, 0.125f);   // * DH^-0.5
  gemm_wmma_f16<<<gemm_blocks, 256, 0, stream>>>(Xk16, Wk16, bk, Kp16, nullptr,
                                                 MT, Dq, Dq, 1.0f);
  gemm_wmma_f16<<<gemm_blocks, 256, 0, stream>>>(Xv16, Wv16, bv, Vp16, nullptr,
                                                 MT, Dq, Dq, 1.0f);
  gemm_wmma_f16<<<gemm_blocks, 256, 0, stream>>>(Xr16, Wr16, nullptr, Rp16, nullptr,
                                                 MT, Dq, Dq, 0.1f);     // fold 0.1 into R

  // 3b) per-head transpose of K and R for contiguous score-phase B loads
  const int tr_blocks = (Tq / 16) * (DHq / 16) * Hq * Bq;  // 16384
  transpose_head_kernel<<<tr_blocks, 256, 0, stream>>>(Kp16, Kt16);
  transpose_head_kernel<<<tr_blocks, 256, 0, stream>>>(Rp16, Rt16);

  // 4) fused attention (TDM V stage + scores + mask + softmax + probs + PV)
  float* out_f32   = (float*)d_out;
  float* probs_f32 = out_f32 + (size_t)MT * Dq;
  (void)hipFuncSetAttribute((const void*)attn_softmax_pv,
                            hipFuncAttributeMaxDynamicSharedMemorySize,
                            SMEM_BYTES);
  attn_softmax_pv<<<Bq * Hq * (Tq / 16), 256, SMEM_BYTES, stream>>>(
      Q16, Kt16, Vp16, Rt16, mask, probs_f32, AO16);

  // 5) output projection -> fp32 d_out
  gemm_wmma_f16<<<gemm_blocks, 256, 0, stream>>>(AO16, Wo16, bo, nullptr, out_f32,
                                                 MT, Dq, Dq, 1.0f);
}
